// QueryLinguishModule_43344809952152
// MI455X (gfx1250) — matrix operand order repaired and checked by reading, not compiled, not run on hardware
//
#include <hip/hip_runtime.h>
#include <hip/hip_bf16.h>
#include <math.h>

// CDNA5 (gfx1250) wave32 WMMA types
typedef __attribute__((ext_vector_type(16))) __bf16 v16bf;
typedef __attribute__((ext_vector_type(8)))  float  v8f;

#define B_   8
#define NQ   4096
#define KK   1024
#define CC   1024

// ---------------------------------------------------------------------------
// Fragment loaders (ISA 7.12.2 layouts, wave32)
//   A 16x32 bf16: lane half h, row r=lane%16; element i -> K = i+(i>=8?8:0)+8h
//     => two contiguous 8-element (16B) chunks at kbase+8h and kbase+16+8h
//   B 32x16 bf16: lane half h, col n=lane%16; element i -> K = i+(i>=8?8:0)+8h
//     => two contiguous 8-element (16B) chunks at kbase+8h and kbase+16+8h (from K-major src)
//   C/D 16x16 f32: vgpr v, lane half h, col r -> row = v+8h
// ---------------------------------------------------------------------------
__device__ __forceinline__ v16bf loadA_f32(const float* p /*rowbase+kbase*/, int h) {
    float t[16];
    *(float4*)(t + 0)  = *(const float4*)(p + 8 * h);
    *(float4*)(t + 4)  = *(const float4*)(p + 8 * h + 4);
    *(float4*)(t + 8)  = *(const float4*)(p + 16 + 8 * h);
    *(float4*)(t + 12) = *(const float4*)(p + 16 + 8 * h + 4);
    v16bf a;
#pragma unroll
    for (int i = 0; i < 16; ++i) a[i] = (__bf16)t[i];
    return a;
}

__device__ __forceinline__ v16bf loadA_bf16(const __bf16* p /*rowbase+kbase*/, int h) {
    union { v16bf v; uint4 q[2]; } u;
    u.q[0] = *(const uint4*)(p + 8 * h);
    u.q[1] = *(const uint4*)(p + 16 + 8 * h);
    return u.v;
}

// B fragment from a K-contiguous (transposed) bf16 source: p = colbase*ld + kbase
__device__ __forceinline__ v16bf loadB_tr(const __bf16* p, int h) {
    union { v16bf v; uint4 q[2]; } u;
    u.q[0] = *(const uint4*)(p + 8 * h);
    u.q[1] = *(const uint4*)(p + 16 + 8 * h);
    return u.v;
}

// Branch-free exact-form gelu: erf via A&S 7.1.26 (|err| < 1.5e-7)
__device__ __forceinline__ float gelu_erf(float x) {
    float z  = x * 0.70710678118654752f;
    float az = fabsf(z);
    float t  = __builtin_amdgcn_rcpf(1.0f + 0.3275911f * az);
    float p  = t * (0.254829592f +
               t * (-0.284496736f +
               t * (1.421413741f +
               t * (-1.453152027f +
               t * 1.061405429f))));
    float e  = __expf(-az * az);
    float er = copysignf(1.0f - p * e, z);
    return 0.5f * x * (1.0f + er);
}

// ---------------------------------------------------------------------------
// K0a: transpose + f32->bf16 convert (LDS-tiled, both sides coalesced)
// ---------------------------------------------------------------------------
__global__ __launch_bounds__(256)
void tconv_kernel(const float* __restrict__ src, __bf16* __restrict__ dst,
                  int rows, int cols) {
    __shared__ __bf16 tile[32][33];
    const int tx = threadIdx.x & 31, ty = threadIdx.x >> 5;  // 32 x 8
    const int c0 = blockIdx.x * 32, r0 = blockIdx.y * 32;
    const size_t slice = (size_t)rows * cols;
    src += (size_t)blockIdx.z * slice;
    dst += (size_t)blockIdx.z * slice;
#pragma unroll
    for (int j = 0; j < 32; j += 8)
        tile[ty + j][tx] = (__bf16)src[(size_t)(r0 + ty + j) * cols + c0 + tx];
    __syncthreads();
#pragma unroll
    for (int j = 0; j < 32; j += 8)
        dst[(size_t)(c0 + ty + j) * rows + r0 + tx] = tile[tx][ty + j];
}

// ---------------------------------------------------------------------------
// K0b: straight f32->bf16 convert, 8 elements per thread (b128 in, b128 out)
// ---------------------------------------------------------------------------
__global__ __launch_bounds__(256)
void conv_kernel(const float* __restrict__ src, __bf16* __restrict__ dst) {
    const size_t i = ((size_t)blockIdx.x * 256 + threadIdx.x) * 8;
    float t[8];
    *(float4*)(t + 0) = *(const float4*)(src + i);
    *(float4*)(t + 4) = *(const float4*)(src + i + 4);
    union { __bf16 h[8]; uint4 q; } u;
#pragma unroll
    for (int j = 0; j < 8; ++j) u.h[j] = (__bf16)t[j];
    *(uint4*)(dst + i) = u.q;
}

// ---------------------------------------------------------------------------
// K1: Abf[b] = bf16( gelu( T[b] @ W ) )   [KKxCC]
//     A from f32 T rows, B from bf16 W^T (K-contiguous). 4x2 tiles/wave.
// ---------------------------------------------------------------------------
__global__ __launch_bounds__(32)
void gelu_gemm_kernel(const float* __restrict__ T, const __bf16* __restrict__ Wt,
                      __bf16* __restrict__ Abf) {
    const int lane = threadIdx.x & 31;
    const int h = lane >> 4, r = lane & 15;
    const int m0 = blockIdx.x * 64;     // K-dim rows
    const int n0 = blockIdx.y * 32;     // c cols
    const int b  = blockIdx.z;
    const float* Tb = T + (size_t)b * KK * CC;

    v8f acc[4][2] = {};
    for (int cc = 0; cc < CC; cc += 32) {
        v16bf a[4], bb[2];
#pragma unroll
        for (int m = 0; m < 4; ++m)
            a[m] = loadA_f32(Tb + (size_t)(m0 + m * 16 + r) * CC + cc, h);
#pragma unroll
        for (int n = 0; n < 2; ++n)
            bb[n] = loadB_tr(Wt + (size_t)(n0 + n * 16 + r) * CC + cc, h);
#pragma unroll
        for (int m = 0; m < 4; ++m)
#pragma unroll
            for (int n = 0; n < 2; ++n)
                acc[m][n] = __builtin_amdgcn_wmma_f32_16x16x32_bf16(
                    false, a[m], false, bb[n], (short)0, acc[m][n], false, false);
    }

    __bf16* dst = Abf + (size_t)b * KK * CC;
#pragma unroll
    for (int m = 0; m < 4; ++m)
#pragma unroll
        for (int n = 0; n < 2; ++n)
#pragma unroll
            for (int v = 0; v < 8; ++v)
                dst[(size_t)(m0 + m * 16 + v + 8 * h) * CC + n0 + n * 16 + r] =
                    (__bf16)gelu_erf(acc[m][n][v]);
}

// ---------------------------------------------------------------------------
// K2/K3 (per batch): C = A (4096x1024 bf16) @ Bt^T (1024x1024 bf16, K-major)
//   4 waves/WG, 128x128 WG tile, 64x64 (4x4 WMMA) per wave.
//   Double-buffered LDS staging: 16 KB/slab, one barrier per k-step.
//   STORE_BF16: true -> bf16 scores (K2), false -> f32 output (K3).
// ---------------------------------------------------------------------------
template <bool STORE_BF16>
__global__ __launch_bounds__(128)
void mm_lds_kernel(const __bf16* __restrict__ A, const __bf16* __restrict__ Bt,
                   void* __restrict__ outv) {
    __shared__ __bf16 As[2][128][32];
    __shared__ __bf16 Bs[2][128][32];

    const int t    = threadIdx.x;
    const int lane = t & 31;
    const int w    = t >> 5;                 // wave 0..3
    const int wr   = w >> 1, wc = w & 1;     // 2x2 wave grid
    const int h = lane >> 4, r = lane & 15;
    const int m0 = blockIdx.x * 128;         // M rows
    const int n0 = blockIdx.y * 128;         // N cols

    // cooperative stage of one 32-wide k-slab: 64B contiguous per thread/side
    auto stage = [&](int buf, int cc) {
        const uint4* ga = (const uint4*)(A  + (size_t)(m0 + t) * 1024 + cc);
        const uint4* gb = (const uint4*)(Bt + (size_t)(n0 + t) * 1024 + cc);
        uint4* la = (uint4*)&As[buf][t][0];
        uint4* lb = (uint4*)&Bs[buf][t][0];
#pragma unroll
        for (int j = 0; j < 4; ++j) { la[j] = ga[j]; lb[j] = gb[j]; }
    };

    stage(0, 0);
    __syncthreads();

    v8f acc[4][4] = {};
    for (int cc = 0; cc < 1024; cc += 32) {
        const int buf = (cc >> 5) & 1;
        if (cc + 32 < 1024) stage(buf ^ 1, cc + 32);

        v16bf a[4], bb[4];
#pragma unroll
        for (int m = 0; m < 4; ++m) {
            const __bf16* p = &As[buf][wr * 64 + m * 16 + r][0];
            union { v16bf vv; uint4 q[2]; } u;
            u.q[0] = *(const uint4*)(p + 8 * h);
            u.q[1] = *(const uint4*)(p + 16 + 8 * h);
            a[m] = u.vv;
        }
#pragma unroll
        for (int n = 0; n < 4; ++n) {
            const __bf16* p = &Bs[buf][wc * 64 + n * 16 + r][0];
            union { v16bf vv; uint4 q[2]; } u;
            u.q[0] = *(const uint4*)(p + 8 * h);
            u.q[1] = *(const uint4*)(p + 16 + 8 * h);
            bb[n] = u.vv;
        }
#pragma unroll
        for (int m = 0; m < 4; ++m)
#pragma unroll
            for (int n = 0; n < 4; ++n)
                acc[m][n] = __builtin_amdgcn_wmma_f32_16x16x32_bf16(
                    false, a[m], false, bb[n], (short)0, acc[m][n], false, false);
        __syncthreads();
    }

#pragma unroll
    for (int m = 0; m < 4; ++m)
#pragma unroll
        for (int n = 0; n < 4; ++n)
#pragma unroll
            for (int v = 0; v < 8; ++v) {
                const size_t idx =
                    (size_t)(m0 + wr * 64 + m * 16 + v + 8 * h) * 1024 +
                    (n0 + wc * 64 + n * 16 + r);
                if constexpr (STORE_BF16)
                    ((__bf16*)outv)[idx] = (__bf16)acc[m][n][v];
                else
                    ((float*)outv)[idx] = acc[m][n][v];
            }
}

// ---------------------------------------------------------------------------
extern "C" void kernel_launch(void* const* d_in, const int* in_sizes, int n_in,
                              void* d_out, int out_size, void* d_ws, size_t ws_size,
                              hipStream_t stream) {
    const float* Q = (const float*)d_in[0];                 // [8,4096,1024] f32
    const float* T = (const float*)d_in[1];                 // [8,1024,1024] f32
    const float* W = (const float*)d_in[2];                 // [1024,1024]   f32
    float* out = (float*)d_out;                             // [8,4096,1024] f32

    // workspace layout (bf16 elements), ~106 MB total:
    //   Ttbf [B, CC, KK]   (T transposed)      8,388,608 elts (16 MB)
    //   Wtbf [CC, CC]      (W transposed)      1,048,576 elts ( 2 MB)
    //   Abf  [B, KK, CC]   (gelu(T@W))         8,388,608 elts (16 MB)
    //   Sbf  [NQ, KK]      (per-batch scores)  4,194,304 elts ( 8 MB)
    //   Qbf  [B, NQ, CC]   (Q in bf16)        33,554,432 elts (64 MB)
    __bf16* Ttbf = (__bf16*)d_ws;
    __bf16* Wtbf = Ttbf + (size_t)B_ * KK * CC;
    __bf16* Abf  = Wtbf + (size_t)CC * CC;
    __bf16* Sbf  = Abf  + (size_t)B_ * KK * CC;
    __bf16* Qbf  = Sbf  + (size_t)NQ * KK;

    // Stage 0: transpose-convert T (per batch) and W; straight-convert Q
    tconv_kernel<<<dim3(CC / 32, KK / 32, B_), 256, 0, stream>>>(T, Ttbf, KK, CC);
    tconv_kernel<<<dim3(CC / 32, CC / 32, 1), 256, 0, stream>>>(W, Wtbf, CC, CC);
    conv_kernel<<<(size_t)B_ * NQ * CC / (256 * 8), 256, 0, stream>>>(Q, Qbf);

    // Stage 1: Abf = bf16(gelu(T @ W)), all batches
    gelu_gemm_kernel<<<dim3(KK / 64, CC / 32, B_), 32, 0, stream>>>(T, Wtbf, Abf);

    // Stages 2+3 per batch: keeps the 8 MB score slab L2-resident
    for (int b = 0; b < B_; ++b) {
        const __bf16* Qb  = Qbf  + (size_t)b * NQ * CC;
        const __bf16* Ab  = Abf  + (size_t)b * KK * CC;
        const __bf16* Ttb = Ttbf + (size_t)b * KK * CC;
        float*        Ob  = out  + (size_t)b * NQ * CC;
        // K2: Sbf = bf16(Qb @ Ab^T)
        mm_lds_kernel<true><<<dim3(NQ / 128, KK / 128), 128, 0, stream>>>(Qb, Ab, Sbf);
        // K3: Ob = Sbf @ Ttb^T (= Sb @ T_b)
        mm_lds_kernel<false><<<dim3(NQ / 128, CC / 128), 128, 0, stream>>>(Sbf, Ttb, Ob);
    }
}
